// MemoryAttentionLayer_21741124452850
// MI455X (gfx1250) — compile-verified, matrix operand order checked by reading, not run on hardware
//
#include <hip/hip_runtime.h>
#include <math.h>

// ---------------- problem constants ----------------
#define B_    4
#define T_    512
#define H_    768
#define M_    1024
#define MS_   131072
#define D_    128
#define KTOP  32
#define NSPLIT 64                      // global key splits
#define KEYS_PER_SPLIT (MS_ / NSPLIT)  // 2048 keys per wave-split
#define LN_EPS 1e-12f

typedef float  v8f  __attribute__((ext_vector_type(8)));
typedef __bf16 v8bf __attribute__((ext_vector_type(8)));
typedef __bf16 v16bf __attribute__((ext_vector_type(16)));

// =====================================================================
// K0: split f32 keys into bf16 hi/lo planes (error-compensated GEMM).
//   x = (float)hi + (float)lo + O(2^-18 * x)
// =====================================================================
__global__ __launch_bounds__(256) void split_keys_kernel(
    const float* __restrict__ keys,
    unsigned int* __restrict__ kH32, unsigned int* __restrict__ kL32) {
  const size_t n2 = (size_t)MS_ * D_ / 2;   // pairs
  for (size_t i = (size_t)blockIdx.x * 256 + threadIdx.x; i < n2;
       i += (size_t)gridDim.x * 256) {
    float x0 = keys[2 * i], x1 = keys[2 * i + 1];
    __bf16 h0 = (__bf16)x0, h1 = (__bf16)x1;
    __bf16 l0 = (__bf16)(x0 - (float)h0), l1 = (__bf16)(x1 - (float)h1);
    union { __bf16 b[2]; unsigned int u; } ph, pl;
    ph.b[0] = h0; ph.b[1] = h1;
    pl.b[0] = l0; pl.b[1] = l1;
    kH32[i] = ph.u; kL32[i] = pl.u;
  }
}

// =====================================================================
// K1: query projection.  8 mentions per block, 128 threads (one per d).
//   queries[m] = concat(enc[b,start], enc[b,end]) @ W_q + b_q   (f32)
// =====================================================================
__global__ __launch_bounds__(128) void qproj_kernel(
    const float* __restrict__ enc,
    const int* __restrict__ mb, const int* __restrict__ msp,
    const int* __restrict__ mep,
    const float* __restrict__ Wq, const float* __restrict__ bq,
    float* __restrict__ q) {
  const int d  = threadIdx.x;      // 0..127
  const int m0 = blockIdx.x * 8;
  __shared__ int sbase[8][2];
  if (threadIdx.x < 16) {
    int j = threadIdx.x >> 1;
    int m = m0 + j;
    int b = mb[m];
    int pos = (threadIdx.x & 1) ? mep[m] : msp[m];
    sbase[j][threadIdx.x & 1] = (b * T_ + pos) * H_;
  }
  __syncthreads();
  float acc[8];
#pragma unroll
  for (int j = 0; j < 8; ++j) acc[j] = 0.f;
  for (int k = 0; k < H_; ++k) {
    float w0 = Wq[(size_t)k * D_ + d];
    float w1 = Wq[(size_t)(H_ + k) * D_ + d];
#pragma unroll
    for (int j = 0; j < 8; ++j)
      acc[j] += enc[sbase[j][0] + k] * w0 + enc[sbase[j][1] + k] * w1;
  }
  float bb = bq[d];
#pragma unroll
  for (int j = 0; j < 8; ++j) q[(size_t)(m0 + j) * D_ + d] = acc[j] + bb;
}

// ---------------------------------------------------------------------
// helpers: build bf16 hi/lo fragments
//   16-bit operand layout (16x32 A, mirrored 32x16 B), lane = (idx, half):
//   elements 0..7  -> K = kk + 8*half + 0..7
//   elements 8..15 -> K = kk + 16 + 8*half + 0..7
// ---------------------------------------------------------------------
__device__ inline void build_frag_f32(const float* __restrict__ p0,
                                      const float* __restrict__ p1,
                                      v16bf& fh, v16bf& fl) {
  v8bf h0, h1, l0, l1;
#pragma unroll
  for (int i = 0; i < 8; ++i) {
    float x = p0[i];
    __bf16 h = (__bf16)x;
    h0[i] = h; l0[i] = (__bf16)(x - (float)h);
    float y = p1[i];
    __bf16 g = (__bf16)y;
    h1[i] = g; l1[i] = (__bf16)(y - (float)g);
  }
  fh = __builtin_shufflevector(h0, h1, 0,1,2,3,4,5,6,7,8,9,10,11,12,13,14,15);
  fl = __builtin_shufflevector(l0, l1, 0,1,2,3,4,5,6,7,8,9,10,11,12,13,14,15);
}

__device__ inline v16bf load_frag_bf16(const __bf16* __restrict__ row,
                                       int off0, int off1) {
  v8bf a = *(const v8bf*)(row + off0);
  v8bf b = *(const v8bf*)(row + off1);
  return __builtin_shufflevector(a, b, 0,1,2,3,4,5,6,7,8,9,10,11,12,13,14,15);
}

// =====================================================================
// K2: fused score GEMM (split-bf16 WMMA 16x16x32) + streaming top-32.
//   scores = Qh.Kh + Qh.Kl + Ql.Kh  (f32 accumulate) ~= f32 GEMM
//   grid = (M/16 mention tiles, NSPLIT/8); block = 256 = 8 waves.
//   Each wave owns one 2048-key split; lanes 0..15 each own one
//   mention's running top-32 list (LDS + register min tracking).
// =====================================================================
__global__ __launch_bounds__(256) void score_topk_kernel(
    const float* __restrict__ q,
    const __bf16* __restrict__ kH, const __bf16* __restrict__ kL,
    float* __restrict__ candS, int* __restrict__ candI) {
  __shared__ float sLs[8 * 16 * 32];      // per-wave per-mention top scores
  __shared__ int   sLi[8 * 16 * 32];      // per-wave per-mention top indices
  __shared__ float sT[8 * 16 * 20];       // per-wave 16x16 score staging (pad 20)

  const int tid   = threadIdx.x;
  const int w     = tid >> 5;
  const int lane  = tid & 31;
  const int half  = lane >> 4;
  const int l15   = lane & 15;
  const int m0    = blockIdx.x * 16;
  const int split = blockIdx.y * 8 + w;   // 0..NSPLIT-1

  // A fragments (queries, hi/lo) for this lane: K = 0..127 in 4 chunks.
  v16bf aH[4], aL[4];
  {
    const float* qrow = q + (size_t)(m0 + l15) * D_;
#pragma unroll
    for (int j = 0; j < 4; ++j) {
      const int kk = 32 * j;
      build_frag_f32(qrow + kk + 8 * half, qrow + kk + 16 + 8 * half,
                     aH[j], aL[j]);
    }
  }

  // init this wave's lists
  float* myLs = &sLs[w * (16 * 32)];
  int*   myLi = &sLi[w * (16 * 32)];
  for (int j = lane; j < 16 * 32; j += 32) { myLs[j] = -__builtin_inff(); myLi[j] = 0; }
  __builtin_amdgcn_wave_barrier();

  float* myT = &sT[w * (16 * 20)];
  float curMin = -__builtin_inff();
  int   minPos = 0;
  const int keyBase = split * KEYS_PER_SPLIT;

  for (int st = 0; st < KEYS_PER_SPLIT / 16; ++st) {
    const int key0 = keyBase + st * 16;
    const __bf16* rowH = kH + (size_t)(key0 + l15) * D_;
    const __bf16* rowL = kL + (size_t)(key0 + l15) * D_;
    v8f c = {0.f, 0.f, 0.f, 0.f, 0.f, 0.f, 0.f, 0.f};
#pragma unroll
    for (int j = 0; j < 4; ++j) {
      const int kk = 32 * j;
      v16bf bH = load_frag_bf16(rowH, kk + 8 * half, kk + 16 + 8 * half);
      v16bf bL = load_frag_bf16(rowL, kk + 8 * half, kk + 16 + 8 * half);
      c = __builtin_amdgcn_wmma_f32_16x16x32_bf16(false, aH[j], false, bH,
                                                  (short)0, c, false, false);
      c = __builtin_amdgcn_wmma_f32_16x16x32_bf16(false, aH[j], false, bL,
                                                  (short)0, c, false, false);
      c = __builtin_amdgcn_wmma_f32_16x16x32_bf16(false, aL[j], false, bH,
                                                  (short)0, c, false, false);
    }
    // stage scores: C layout -> mention row (r + 8*half), key col l15
#pragma unroll
    for (int r = 0; r < 8; ++r) myT[(r + 8 * half) * 20 + l15] = c[r];
    __builtin_amdgcn_wave_barrier();
    if (lane < 16) {                       // lane t owns mention m0+t
      for (int n = 0; n < 16; ++n) {
        float s = myT[lane * 20 + n];
        if (s > curMin) {
          myLs[lane * 32 + minPos] = s;
          myLi[lane * 32 + minPos] = key0 + n;
          float mn = myLs[lane * 32 + 0]; int mp = 0;
          for (int j = 1; j < 32; ++j) {
            float v = myLs[lane * 32 + j];
            if (v < mn) { mn = v; mp = j; }
          }
          curMin = mn; minPos = mp;
        }
      }
    }
    __builtin_amdgcn_wave_barrier();
  }
  // flush this wave's lists to the candidate buffer
  if (lane < 16) {
    size_t base = ((size_t)(m0 + lane) * NSPLIT + split) * KTOP;
    for (int j = 0; j < KTOP; ++j) {
      candS[base + j] = myLs[lane * 32 + j];
      candI[base + j] = myLi[lane * 32 + j];
    }
  }
}

// =====================================================================
// K3: per mention: merge NSPLIT*32 candidates -> top-32, softmax,
//     weighted value gather (f32 keys), update projection (x mask).
// =====================================================================
__global__ __launch_bounds__(256) void merge_retrieve_kernel(
    const float* __restrict__ candS, const int* __restrict__ candI,
    const float* __restrict__ keys, const float* __restrict__ Wu,
    const float* __restrict__ bu, const float* __restrict__ mask,
    float* __restrict__ upd) {
  __shared__ float cs[NSPLIT * KTOP];
  __shared__ int   ci[NSPLIT * KTOP];
  __shared__ float rmax[256];
  __shared__ int   rpos[256];
  __shared__ float topS[KTOP];
  __shared__ int   topI[KTOP];
  __shared__ float attn[KTOP];
  __shared__ float retr[D_];
  const int tid = threadIdx.x;
  const int m   = blockIdx.x;
  const size_t cbase = (size_t)m * (NSPLIT * KTOP);
  for (int j = tid; j < NSPLIT * KTOP; j += 256) {
    cs[j] = candS[cbase + j];
    ci[j] = candI[cbase + j];
  }
  __syncthreads();
  // iterative parallel argmax extraction (descending)
  for (int t = 0; t < KTOP; ++t) {
    float mx = -__builtin_inff(); int mp = -1;
    for (int j = tid; j < NSPLIT * KTOP; j += 256) {
      float v = cs[j];
      if (v > mx) { mx = v; mp = j; }
    }
    rmax[tid] = mx; rpos[tid] = mp;
    __syncthreads();
    for (int s = 128; s > 0; s >>= 1) {
      if (tid < s && rmax[tid + s] > rmax[tid]) {
        rmax[tid] = rmax[tid + s]; rpos[tid] = rpos[tid + s];
      }
      __syncthreads();
    }
    if (tid == 0) {
      topS[t] = rmax[0];
      topI[t] = (rpos[0] >= 0) ? ci[rpos[0]] : 0;
      if (rpos[0] >= 0) cs[rpos[0]] = -__builtin_inff();
    }
    __syncthreads();
  }
  // softmax over top scores (topS[0] is the max)
  if (tid < KTOP) attn[tid] = __expf(topS[tid] - topS[0]);
  __syncthreads();
  if (tid == 0) {
    float s = 0.f;
    for (int j = 0; j < KTOP; ++j) s += attn[j];
    rmax[0] = 1.f / s;
  }
  __syncthreads();
  // retrieved[d] = sum_k attn[k] * keys[topI[k]][d]
  if (tid < D_) {
    float inv = rmax[0];
    float acc = 0.f;
    for (int k = 0; k < KTOP; ++k)
      acc += attn[k] * keys[(size_t)topI[k] * D_ + tid];
    retr[tid] = acc * inv;
  }
  __syncthreads();
  // upd = (retr @ W_u + b_u) * mask[m]
  const float mk = mask[m];
  for (int h = tid; h < H_; h += 256) {
    float acc = bu[h];
    for (int d = 0; d < D_; ++d) acc += retr[d] * Wu[(size_t)d * H_ + h];
    upd[(size_t)m * H_ + h] = acc * mk;
  }
}

// =====================================================================
// K4: scatter-add upd rows at (b,start) and (b,end) into out.
// =====================================================================
__global__ __launch_bounds__(256) void scatter_kernel(
    const float* __restrict__ upd,
    const int* __restrict__ mb, const int* __restrict__ msp,
    const int* __restrict__ mep, float* __restrict__ out) {
  const int m = blockIdx.x;
  const int b = mb[m], s = msp[m], e = mep[m];
  const size_t sb = ((size_t)b * T_ + s) * H_;
  const size_t eb = ((size_t)b * T_ + e) * H_;
  for (int h = threadIdx.x; h < H_; h += 256) {
    float v = upd[(size_t)m * H_ + h];
    atomicAdd(&out[sb + h], v);
    atomicAdd(&out[eb + h], v);
  }
}

// =====================================================================
// K5: in-place row LayerNorm over hidden dim.
// =====================================================================
__global__ __launch_bounds__(256) void ln_kernel(
    float* __restrict__ out, const float* __restrict__ g,
    const float* __restrict__ bta) {
  __shared__ float red[256];
  const int row = blockIdx.x;
  float* p = out + (size_t)row * H_;
  const int tid = threadIdx.x;
  float s = 0.f;
  for (int h = tid; h < H_; h += 256) s += p[h];
  red[tid] = s; __syncthreads();
  for (int st = 128; st > 0; st >>= 1) {
    if (tid < st) red[tid] += red[tid + st];
    __syncthreads();
  }
  const float mean = red[0] / (float)H_;
  __syncthreads();
  float v = 0.f;
  for (int h = tid; h < H_; h += 256) { float d = p[h] - mean; v += d * d; }
  red[tid] = v; __syncthreads();
  for (int st = 128; st > 0; st >>= 1) {
    if (tid < st) red[tid] += red[tid + st];
    __syncthreads();
  }
  const float inv = rsqrtf(red[0] / (float)H_ + LN_EPS);
  for (int h = tid; h < H_; h += 256) p[h] = (p[h] - mean) * inv * g[h] + bta[h];
}

// =====================================================================
extern "C" void kernel_launch(void* const* d_in, const int* in_sizes, int n_in,
                              void* d_out, int out_size, void* d_ws, size_t ws_size,
                              hipStream_t stream) {
  const float* enc  = (const float*)d_in[0];
  const int*   mb   = (const int*)d_in[1];
  const int*   msp  = (const int*)d_in[2];
  const int*   mep  = (const int*)d_in[3];
  const float* mask = (const float*)d_in[4];
  const float* keys = (const float*)d_in[5];   // [131072,128] f32
  // d_in[6] memory_identifiers, d_in[7] memory_entity_ids: unused by reference output
  const float* Wq = (const float*)d_in[8];
  const float* bq = (const float*)d_in[9];
  const float* Wu = (const float*)d_in[10];
  const float* bu = (const float*)d_in[11];
  const float* lng = (const float*)d_in[12];
  const float* lnb = (const float*)d_in[13];
  float* out = (float*)d_out;

  // workspace layout
  char* ws = (char*)d_ws;
  float* qbuf = (float*)ws;                                   // M*D f32
  size_t off  = (size_t)M_ * D_ * sizeof(float);
  __bf16* kH  = (__bf16*)(ws + off);                          // MS*D bf16
  off += (size_t)MS_ * D_ * sizeof(__bf16);
  __bf16* kL  = (__bf16*)(ws + off);                          // MS*D bf16
  off += (size_t)MS_ * D_ * sizeof(__bf16);
  float* candS = (float*)(ws + off);                          // M*NSPLIT*KTOP
  off += (size_t)M_ * NSPLIT * KTOP * sizeof(float);
  int*   candI = (int*)(ws + off);                            // M*NSPLIT*KTOP
  off += (size_t)M_ * NSPLIT * KTOP * sizeof(int);
  float* upd   = (float*)(ws + off);                          // M*H

  split_keys_kernel<<<4096, 256, 0, stream>>>(keys, (unsigned int*)kH,
                                              (unsigned int*)kL);

  qproj_kernel<<<M_ / 8, 128, 0, stream>>>(enc, mb, msp, mep, Wq, bq, qbuf);

  dim3 g2(M_ / 16, NSPLIT / 8);
  score_topk_kernel<<<g2, 256, 0, stream>>>(qbuf, kH, kL, candS, candI);

  merge_retrieve_kernel<<<M_, 256, 0, stream>>>(candS, candI, keys, Wu, bu, mask, upd);

  hipMemcpyAsync(out, enc, (size_t)B_ * T_ * H_ * sizeof(float),
                 hipMemcpyDeviceToDevice, stream);

  scatter_kernel<<<M_, 256, 0, stream>>>(upd, mb, msp, mep, out);

  ln_kernel<<<B_ * T_, 256, 0, stream>>>(out, lng, lnb);
}
